// VariableBandWindowAttention_85770496901699
// MI455X (gfx1250) — matrix operand order repaired and checked by reading, not compiled, not run on hardware
//
#include <hip/hip_runtime.h>

typedef __attribute__((ext_vector_type(16))) __bf16 v16bf;
typedef __attribute__((ext_vector_type(8)))  float  v8f;
typedef __attribute__((ext_vector_type(4)))  unsigned tdm_g0_t;
typedef __attribute__((ext_vector_type(8)))  int      tdm_g1_t;
typedef __attribute__((ext_vector_type(4)))  int      tdm_g23_t;

// problem constants
constexpr int B_  = 8192;
constexpr int N_  = 49;
constexpr int C_  = 192;
constexpr int H_  = 6;
constexpr int HD_ = 32;
constexpr int NP  = 64;     // tokens padded to multiple of 16
constexpr int THREADS = 256;
constexpr int WAVES = 8;
constexpr float SCALE = 0.17677669529663687f;   // 1/sqrt(32)

// padded LDS row strides (elements; even, odd-dword for bank spread)
constexpr int XST = 194;    // xs / o_s   [NP][C_]  rows        (97 dwords)
constexpr int QST = 34;     // q_s / k_s  [H][NP][HD] token rows (17 dwords)
constexpr int VST = 66;     // v_s        [H][HD][NP] token stride (33 dwords)
constexpr int PST = 66;     // p_s        [H][NP][NP] key stride   (33 dwords)

constexpr int XS_E = NP * XST;            // 12416
constexpr int QS_E = H_ * NP * QST;       // 13056
constexpr int VS_E = H_ * HD_ * VST;      // 12672
constexpr int PS_E = H_ * NP * PST;       // 25344
constexpr int SMEM_ELEMS = XS_E + 2 * QS_E + VS_E + PS_E;   // 76544 -> 149.5 KB

__device__ __forceinline__ v8f wmma_bf16(v16bf a, v16bf b, v8f c) {
  return __builtin_amdgcn_wmma_f32_16x16x32_bf16(false, a, false, b, (short)0, c,
                                                 false, false);
}

// A fragment: 16x32 bf16 tile, row-major in LDS (contiguous k: lane half h holds k = 16h..16h+15).
__device__ __forceinline__ v16bf load_a_lds(const __bf16* base, int stride, int lane) {
  const int row = lane & 15;
  const int kb2 = (lane >> 4) << 3;             // 0 or 8 (uint index)
  union { v16bf v; unsigned u[8]; } r;
  const unsigned* p = (const unsigned*)(base + row * stride);
  #pragma unroll
  for (int j = 0; j < 4; ++j) r.u[j]     = p[kb2 + j];
  #pragma unroll
  for (int j = 0; j < 4; ++j) r.u[4 + j] = p[4 + kb2 + j];
  return r.v;
}

// B fragment (32x16) from an N-major LDS array arr[n][k]: 16 contiguous halves.
__device__ __forceinline__ v16bf load_bt_lds(const __bf16* base, int stride, int lane) {
  const int n  = lane & 15;
  const int kb = (lane >> 4) << 4;              // 0 or 16
  union { v16bf v; unsigned u[8]; } r;
  const unsigned* p = (const unsigned*)(base + n * stride + kb);
  #pragma unroll
  for (int j = 0; j < 8; ++j) r.u[j] = p[j];
  return r.v;
}

// B fragment (32x16) from a global N-major bf16 weight matrix (ld = 192, k0
// multiple of 32): two b128 loads of 16 contiguous halves, no conversion.
__device__ __forceinline__ v16bf load_bt_gl(const __bf16* __restrict__ wt,
                                            int k0, int n0, int lane) {
  const int n  = n0 + (lane & 15);
  const int kb = k0 + ((lane >> 4) << 4);
  union { v16bf v; uint4 q[2]; } r;
  const uint4* p = reinterpret_cast<const uint4*>(wt + (size_t)n * C_ + kb);
  r.q[0] = p[0];
  r.q[1] = p[1];
  return r.v;
}

// one-time pre-pass: transpose + convert both weight matrices to bf16 N-major
__global__ __launch_bounds__(256)
void convert_weights(const float* __restrict__ wq, const float* __restrict__ wp,
                     __bf16* __restrict__ wtq, __bf16* __restrict__ wtp) {
  const int i = blockIdx.x * 256 + threadIdx.x;
  if (i < 3 * C_ * C_) {                        // wtq[n][k] = wq[k][n]
    const int n = i / C_, k = i % C_;
    wtq[i] = (__bf16)wq[(size_t)k * (3 * C_) + n];
  } else if (i < 3 * C_ * C_ + C_ * C_) {       // wtp[n][k] = wp[k][n]
    const int j = i - 3 * C_ * C_;
    const int n = j / C_, k = j % C_;
    wtp[j] = (__bf16)wp[(size_t)k * C_ + n];
  }
}

// second launch-bounds arg (HIP): min waves per EU -> cap VGPRs so 4 waves/SIMD
// stay resident; stops the compiler hoisting 192 VGPRs of A fragments.
__global__ __launch_bounds__(THREADS, 4)
void win_attn_fused(const float* __restrict__ x, const int* __restrict__ mask,
                    const __bf16* __restrict__ wtq, const __bf16* __restrict__ wtp,
                    const float* __restrict__ b_qkv, const float* __restrict__ b_proj,
                    float* __restrict__ out) {
  extern __shared__ char smem_raw[];
  __bf16* xs  = (__bf16*)smem_raw;              // [NP][XST]; o_s aliases (xs dead after ph1)
  __bf16* o_s = xs;
  __bf16* q_s = xs  + XS_E;                     // [H][NP][QST]
  __bf16* k_s = q_s + QS_E;                     // [H][NP][QST]
  __bf16* v_s = k_s + QS_E;                     // [H][HD][VST]   d-major transposed
  __bf16* p_s = v_s + VS_E;                     // [H][NP][PST]
  float*  stage = (float*)(smem_raw + (size_t)(XS_E + 2 * QS_E + VS_E) * 2); // aliases p_s

  const int b    = blockIdx.x;
  const int tid  = threadIdx.x;
  const int lane = tid & 31;
  const int wave = tid >> 5;

  const float* xb = x + (size_t)b * N_ * C_;
  const int*   mb = mask + (size_t)b * N_;

#if __has_builtin(__builtin_amdgcn_tensor_load_to_lds) && \
    __has_builtin(__builtin_amdgcn_s_wait_tensorcnt)
  // ---- stage x via Tensor Data Mover: 2D tensor {192,49}, tile {192,64};
  //      TDM OOB reads return zero -> rows 49..63 zero-padded for free ----
  if (wave == 0) {
    const unsigned lds_addr = (unsigned)(unsigned long long)(const void*)stage;
    const unsigned long long ga = (unsigned long long)(const void*)xb;
    tdm_g0_t g0 = { 1u,                                 // count=1 valid descriptor
                    lds_addr,                           // D#.lds_addr
                    (unsigned)ga,                       // global_addr[31:0] (bits 95:64)
                    (unsigned)(ga >> 32) | (2u << 30) };// global_addr[56:32] | type=2
    tdm_g1_t g1 = { 0x20000,                            // data_size=2 (4B), no multicast
                    C_ << 16,                           // tensor_dim0 low16 (bits 63:48)
                    N_ << 16,                           // tensor_dim1 low16 (bits 95:80)
                    C_ << 16,                           // tile_dim0 (bits 127:112)
                    NP,                                 // tile_dim1 (bits 143:128)
                    C_,                                 // tensor_dim0_stride low32
                    0, 0 };
    tdm_g23_t gz = { 0, 0, 0, 0 };
    tdm_g1_t  gz8 = { 0, 0, 0, 0, 0, 0, 0, 0 };
    __builtin_amdgcn_tensor_load_to_lds(g0, g1, gz, gz, gz8, 0);
    __builtin_amdgcn_s_wait_tensorcnt(0);
  }
  __syncthreads();
  // convert staged f32 -> bf16 with padded row stride
  for (int i = tid; i < NP * C_; i += THREADS) {
    const int r = i / C_, c = i % C_;
    xs[r * XST + c] = (__bf16)stage[i];
  }
#else
  for (int i = tid; i < NP * C_; i += THREADS) {
    const int r = i / C_, c = i % C_;
    xs[r * XST + c] = (__bf16)((r < N_) ? xb[r * C_ + c] : 0.0f);
  }
#endif
  __syncthreads();

  // ---- phase 1: qkv = x @ w_qkv + b_qkv.  Wave owns an n-stripe; each B
  //      fragment loaded ONCE per block (2x b128) and reused across 4 m-tiles.
  #pragma unroll 1
  for (int nt = wave; nt < 36; nt += WAVES) {
    const int n0 = nt * 16;
    v8f acc[4] = {{}, {}, {}, {}};
    #pragma unroll
    for (int kt = 0; kt < 6; ++kt) {
      v16bf bw = load_bt_gl(wtq, kt * 32, n0, lane);
      #pragma unroll
      for (int mt = 0; mt < 4; ++mt) {
        v16bf a = load_a_lds(xs + mt * 16 * XST + kt * 32, XST, lane);
        acc[mt] = wmma_bf16(a, bw, acc[mt]);
      }
    }
    const int col = n0 + (lane & 15);
    const int sel = col / C_;                   // 0=q 1=k 2=v (uniform per stripe)
    const int cc  = col % C_;
    const int h = cc / HD_, d = cc % HD_;
    const float bias = b_qkv[col];
    #pragma unroll
    for (int mt = 0; mt < 4; ++mt) {
      #pragma unroll
      for (int r = 0; r < 8; ++r) {
        const int m = mt * 16 + r + ((lane >> 4) << 3);
        const float val = acc[mt][r] + bias;
        if (sel == 0)      q_s[(h * NP + m) * QST + d] = (__bf16)(val * SCALE);
        else if (sel == 1) k_s[(h * NP + m) * QST + d] = (__bf16)val;
        else               v_s[(h * HD_ + d) * VST + m] = (__bf16)val;   // transposed
      }
    }
  }
  __syncthreads();

  // ---- phase 2: S = q @ k^T, mask, softmax (wave owns 16 rows x 64 keys) ----
  #pragma unroll 1
  for (int rb = wave; rb < H_ * 4; rb += WAVES) {
    const int h = rb >> 2, mt = rb & 3;
    v16bf a = load_a_lds(q_s + (h * NP + mt * 16) * QST, QST, lane);  // K=HD=32
    v8f s[4];
    float mbias[4];
    #pragma unroll
    for (int nt = 0; nt < 4; ++nt) {
      v16bf kf = load_bt_lds(k_s + (h * NP + nt * 16) * QST, QST, lane);
      v8f z = {};
      s[nt] = wmma_bf16(a, kf, z);
      const int key = nt * 16 + (lane & 15);
      mbias[nt] = (key < N_ && mb[key] != 0) ? 0.0f : -__builtin_inff();
    }
    const int rowbase = mt * 16 + ((lane >> 4) << 3);
    #pragma unroll
    for (int r = 0; r < 8; ++r) {
      float sv[4], mx = -__builtin_inff();
      #pragma unroll
      for (int nt = 0; nt < 4; ++nt) { sv[nt] = s[nt][r] + mbias[nt]; mx = fmaxf(mx, sv[nt]); }
      #pragma unroll
      for (int off = 1; off < 16; off <<= 1) mx = fmaxf(mx, __shfl_xor(mx, off, 32));
      float sum = 0.0f;
      #pragma unroll
      for (int nt = 0; nt < 4; ++nt) { sv[nt] = __expf(sv[nt] - mx); sum += sv[nt]; }
      #pragma unroll
      for (int off = 1; off < 16; off <<= 1) sum += __shfl_xor(sum, off, 32);
      const float inv = 1.0f / sum;
      const int row = rowbase + r;
      #pragma unroll
      for (int nt = 0; nt < 4; ++nt)
        p_s[(h * NP + row) * PST + nt * 16 + (lane & 15)] = (__bf16)(sv[nt] * inv);
    }
  }
  __syncthreads();

  // ---- phase 3: o = P @ V; wave owns a (h,dt) stripe, V frag reused x4 ----
  #pragma unroll 1
  for (int s = wave; s < H_ * 2; s += WAVES) {
    const int h = s >> 1, dt = s & 1;
    v8f acc[4] = {{}, {}, {}, {}};
    #pragma unroll
    for (int kt = 0; kt < 2; ++kt) {
      v16bf vf = load_bt_lds(v_s + (h * HD_ + dt * 16) * VST + kt * 32, VST, lane);
      #pragma unroll
      for (int mt = 0; mt < 4; ++mt) {
        v16bf a = load_a_lds(p_s + (h * NP + mt * 16) * PST + kt * 32, PST, lane);
        acc[mt] = wmma_bf16(a, vf, acc[mt]);
      }
    }
    const int col = h * HD_ + dt * 16 + (lane & 15);
    #pragma unroll
    for (int mt = 0; mt < 4; ++mt)
      #pragma unroll
      for (int r = 0; r < 8; ++r) {
        const int m = mt * 16 + r + ((lane >> 4) << 3);
        o_s[m * XST + col] = (__bf16)acc[mt][r];
      }
  }
  __syncthreads();

  // ---- phase 4: y = o @ w_proj + b_proj; same n-stripe / B-reuse scheme ----
  float* yb = out + (size_t)b * N_ * C_;
  #pragma unroll 1
  for (int nt = wave; nt < 12; nt += WAVES) {
    const int n0 = nt * 16;
    v8f acc[4] = {{}, {}, {}, {}};
    #pragma unroll
    for (int kt = 0; kt < 6; ++kt) {
      v16bf bw = load_bt_gl(wtp, kt * 32, n0, lane);
      #pragma unroll
      for (int mt = 0; mt < 4; ++mt) {
        v16bf a = load_a_lds(o_s + mt * 16 * XST + kt * 32, XST, lane);
        acc[mt] = wmma_bf16(a, bw, acc[mt]);
      }
    }
    const int col = n0 + (lane & 15);
    const float bias = b_proj[col];
    #pragma unroll
    for (int mt = 0; mt < 4; ++mt)
      #pragma unroll
      for (int r = 0; r < 8; ++r) {
        const int m = mt * 16 + r + ((lane >> 4) << 3);
        if (m < N_) yb[m * C_ + col] = acc[mt][r] + bias;
      }
  }
}

extern "C" void kernel_launch(void* const* d_in, const int* in_sizes, int n_in,
                              void* d_out, int out_size, void* d_ws, size_t ws_size,
                              hipStream_t stream) {
  const float* x      = (const float*)d_in[0];
  const int*   mask   = (const int*)d_in[1];
  const float* w_qkv  = (const float*)d_in[2];
  const float* b_qkv  = (const float*)d_in[3];
  const float* w_proj = (const float*)d_in[4];
  const float* b_proj = (const float*)d_in[5];
  float* out = (float*)d_out;

  __bf16* wtq = (__bf16*)d_ws;                  // 576x192 bf16, N-major
  __bf16* wtp = wtq + 3 * C_ * C_;              // 192x192 bf16, N-major

  const int conv_elems = 3 * C_ * C_ + C_ * C_;
  convert_weights<<<(conv_elems + 255) / 256, 256, 0, stream>>>(w_qkv, w_proj, wtq, wtp);

  const size_t smem_bytes = (size_t)SMEM_ELEMS * 2;
  win_attn_fused<<<B_, THREADS, smem_bytes, stream>>>(x, mask, wtq, wtp,
                                                      b_qkv, b_proj, out);
}
